// SelfAttention_9388798509324
// MI455X (gfx1250) — matrix operand order repaired and checked by reading, not compiled, not run on hardware
//
#include <hip/hip_runtime.h>

#define BB 8
#define NN 2048
#define DD 512

typedef __attribute__((ext_vector_type(16))) __bf16        v16bf;
typedef __attribute__((ext_vector_type(8)))  __bf16        v8bf;
typedef __attribute__((ext_vector_type(2)))  __bf16        bf16x2;
typedef __attribute__((ext_vector_type(8)))  float         v8f;
typedef __attribute__((ext_vector_type(4)))  unsigned int  u32x4;
typedef __attribute__((ext_vector_type(2)))  float         f32x2;

union U32x8 { unsigned int u[8]; u32x4 q[2]; v16bf bf; v8bf h[2]; };

// packed fp32 pair -> bf16 pair (lowers to one v_cvt_pk_bf16_f32)
__device__ __forceinline__ unsigned int pack2bf(float lo, float hi) {
  f32x2 f; f.x = lo; f.y = hi;
  union { bf16x2 h; unsigned int u; } r;
  r.h = __builtin_convertvector(f, bf16x2);
  return r.u;
}
__device__ __forceinline__ unsigned short f2bf(float f) {
  union { __bf16 h; unsigned short u; } r;
  r.h = (__bf16)f;
  return r.u;
}

// ---------------- prep: W fp32 -> bf16 (2 elems/thread, packed cvt) ----------------
__global__ __launch_bounds__(256) void k_cvt_w(const float* __restrict__ W,
                                               unsigned int* __restrict__ Wb32) {
  int i = blockIdx.x * 256 + threadIdx.x;   // grid == 512*512/(256*2)
  f32x2 f = *(const f32x2*)(W + 2 * (size_t)i);
  Wb32[i] = pack2bf(f.x, f.y);
}

// ---------------- prep: Vt[b][d][n] = bf16(input[b][n][d]) ----------------
__global__ __launch_bounds__(256) void k_transpose_v(const float* __restrict__ inp,
                                                     unsigned short* __restrict__ Vt) {
  __shared__ float tile[32][33];
  int nblk = blockIdx.x % (NN / 32);
  int rest = blockIdx.x / (NN / 32);
  int dblk = rest % (DD / 32);
  int b    = rest / (DD / 32);
  int tx = threadIdx.x & 31, ty = threadIdx.x >> 5;     // 32 x 8
  int n0 = nblk * 32, d0 = dblk * 32;
#pragma unroll
  for (int i = 0; i < 4; ++i) {
    int rr = ty + i * 8;
    tile[rr][tx] = inp[((size_t)(b * NN + n0 + rr)) * DD + d0 + tx];
  }
  __syncthreads();
#pragma unroll
  for (int i = 0; i < 4; ++i) {
    int rr = ty + i * 8;
    Vt[((size_t)(b * DD + d0 + rr)) * NN + n0 + tx] = f2bf(tile[tx][rr]);
  }
}

// ---------------- projection GEMM: Xb = bf16(input @ W^T + b) ----------------
__global__ __launch_bounds__(256) void k_proj(const float* __restrict__ inp,
                                              const unsigned short* __restrict__ Wb,
                                              const float* __restrict__ bias,
                                              unsigned short* __restrict__ Xb) {
  int w = threadIdx.x >> 5, lane = threadIdx.x & 31;
  int tid = blockIdx.x * 8 + w;           // one 16x16 tile per wave
  int mi = tid >> 5, hi = tid & 31;       // 32 h-tiles per row block
  int m0 = mi * 16, h0 = hi * 16;
  int col = lane & 15, kh = lane >> 4, kb = kh * 8;

  v8f c0 = {}, c1 = {};                   // two independent WMMA chains
  const float* arow = inp + (size_t)(m0 + col) * DD;            // A row M == lane&15
  const unsigned short* brow = Wb + (size_t)(h0 + col) * DD + kh * 8;

#pragma unroll 2
  for (int cI = 0; cI < DD / 32; cI += 2) {
#pragma unroll
    for (int half = 0; half < 2; ++half) {
      int k0 = (cI + half) * 32;
      U32x8 ua;
      v8f a0 = *(const v8f*)(arow + k0 + kb);        // 32B, clause of 2x b128
      v8f a1 = *(const v8f*)(arow + k0 + kb + 16);
      ua.h[0] = __builtin_convertvector(a0, v8bf);   // 4x v_cvt_pk_bf16_f32
      ua.h[1] = __builtin_convertvector(a1, v8bf);
      U32x8 ub;
      ub.q[0] = *(const u32x4*)(brow + k0);
      ub.q[1] = *(const u32x4*)(brow + k0 + 16);
      if (half == 0)
        c0 = __builtin_amdgcn_wmma_f32_16x16x32_bf16(false, ua.bf, false, ub.bf,
                                                     (short)0, c0, false, false);
      else
        c1 = __builtin_amdgcn_wmma_f32_16x16x32_bf16(false, ua.bf, false, ub.bf,
                                                     (short)0, c1, false, false);
    }
  }
  v8f c = c0 + c1;
  float bv = bias[h0 + col];
#pragma unroll
  for (int v = 0; v < 8; ++v) {
    int row = v + kh * 8;
    Xb[(size_t)(m0 + row) * DD + h0 + col] = f2bf(c[v] + bv);
  }
}

// ---------------- fused flash attention: softmax(x x^T) @ input ----------------
__global__ __launch_bounds__(256) void k_attn(const unsigned short* __restrict__ Xb,
                                              const unsigned short* __restrict__ Vt,
                                              float* __restrict__ out) {
  __shared__ __align__(16) unsigned short xq[16 * DD];   // 16 KB query tile (bf16)
  __shared__ float S[16][128];                           // 8 KB raw scores
  __shared__ __align__(16) unsigned short P[16][128];    // 4 KB probs (bf16)
  __shared__ float pmax[16][16], psum[16][16];
  __shared__ float mrow[16], lrow[16], srow[16];

  int t = threadIdx.x;
  int w = t >> 5, lane = t & 31;
  int b  = blockIdx.x >> 7;           // NN/16 = 128 query tiles per batch
  int qt = blockIdx.x & 127;
  int q0 = qt * 16;

  // stage the 16 consecutive query rows (contiguous 16 KB) into LDS
  {
    const u32x4* src = (const u32x4*)(Xb + ((size_t)(b * NN + q0)) * DD);
    u32x4* dst = (u32x4*)xq;
#pragma unroll
    for (int i = 0; i < 4; ++i) dst[t + i * 256] = src[t + i * 256];
  }
  if (t < 16) { mrow[t] = -3.0e38f; lrow[t] = 0.0f; }
  __syncthreads();

  int M = lane & 15, kh = lane >> 4, kb = kh * 8;
  int d0 = w * 64;                                   // 64 output cols per wave
  v8f oacc[4] = {v8f{}, v8f{}, v8f{}, v8f{}};

  const unsigned short* xkbase = Xb + (size_t)b * NN * DD;
  const unsigned short* vtb    = Vt + (size_t)b * DD * NN;

  for (int kbk = 0; kbk < NN / 128; ++kbk) {
    int key0 = kbk * 128;
    // warm L2->WGP path for the next key block's B rows
    if (kbk + 1 < NN / 128)
      __builtin_prefetch(xkbase + (size_t)(key0 + 128 + w * 16 + M) * DD, 0, 0);

    // ---- scores: this wave's 16 keys vs the 16 queries (2 WMMA chains) ----
    v8f cs0 = {}, cs1 = {};
    const unsigned short* brow = xkbase + (size_t)(key0 + w * 16 + M) * DD + kh * 8;
#pragma unroll 2
    for (int cI = 0; cI < DD / 32; cI += 2) {
#pragma unroll
      for (int half = 0; half < 2; ++half) {
        int k0 = (cI + half) * 32;
        U32x8 ua;
        ua.q[0] = *(const u32x4*)&xq[M * DD + k0 + kb];
        ua.q[1] = *(const u32x4*)&xq[M * DD + k0 + kb + 16];
        U32x8 ub;
        ub.q[0] = *(const u32x4*)(brow + k0);
        ub.q[1] = *(const u32x4*)(brow + k0 + 16);
        if (half == 0)
          cs0 = __builtin_amdgcn_wmma_f32_16x16x32_bf16(false, ua.bf, false, ub.bf,
                                                        (short)0, cs0, false, false);
        else
          cs1 = __builtin_amdgcn_wmma_f32_16x16x32_bf16(false, ua.bf, false, ub.bf,
                                                        (short)0, cs1, false, false);
      }
    }
    v8f cs = cs0 + cs1;
#pragma unroll
    for (int v = 0; v < 8; ++v)
      S[v + kh * 8][w * 16 + M] = cs[v];
    __syncthreads();

    // ---- block softmax stats (online) ----
    int r = t >> 4, cg = t & 15;
    float lm = -3.0e38f;
#pragma unroll
    for (int j = 0; j < 8; ++j) lm = fmaxf(lm, S[r][cg * 8 + j]);
    pmax[r][cg] = lm;
    __syncthreads();
    if (t < 16) {
      float bm = pmax[t][0];
      for (int j = 1; j < 16; ++j) bm = fmaxf(bm, pmax[t][j]);
      float mold = mrow[t];
      float mnew = fmaxf(mold, bm);
      mrow[t] = mnew;
      srow[t] = __expf(mold - mnew);
    }
    __syncthreads();
    float mn = mrow[r];
    float ls = 0.0f;
    float pv[8];
#pragma unroll
    for (int j = 0; j < 8; ++j) {
      pv[j] = __expf(S[r][cg * 8 + j] - mn);
      ls += pv[j];
    }
    {
      unsigned int* Pd = (unsigned int*)&P[r][cg * 8];
      Pd[0] = pack2bf(pv[0], pv[1]);
      Pd[1] = pack2bf(pv[2], pv[3]);
      Pd[2] = pack2bf(pv[4], pv[5]);
      Pd[3] = pack2bf(pv[6], pv[7]);
    }
    psum[r][cg] = ls;
    __syncthreads();
    if (t < 16) {
      float ssum = 0.0f;
      for (int j = 0; j < 16; ++j) ssum += psum[t][j];
      lrow[t] = lrow[t] * srow[t] + ssum;
    }
    // ---- rescale running output accumulators ----
    float sc[8];
#pragma unroll
    for (int v = 0; v < 8; ++v) sc[v] = srow[v + kh * 8];
#pragma unroll
    for (int nt = 0; nt < 4; ++nt)
#pragma unroll
      for (int v = 0; v < 8; ++v) oacc[nt][v] *= sc[v];

    // ---- O += P @ V  (4 independent WMMA chains, one per output tile) ----
    for (int c4 = 0; c4 < 4; ++c4) {
      U32x8 ua;
      ua.q[0] = *(const u32x4*)&P[M][c4 * 32 + kb];
      ua.q[1] = *(const u32x4*)&P[M][c4 * 32 + kb + 16];
      const unsigned short* vrow = vtb + (size_t)(d0 + M) * NN + key0 + c4 * 32 + kh * 8;
#pragma unroll
      for (int nt = 0; nt < 4; ++nt) {
        U32x8 ub;
        const unsigned short* vp = vrow + (size_t)nt * 16 * NN;
        ub.q[0] = *(const u32x4*)(vp);
        ub.q[1] = *(const u32x4*)(vp + 16);
        oacc[nt] = __builtin_amdgcn_wmma_f32_16x16x32_bf16(false, ua.bf, false, ub.bf,
                                                           (short)0, oacc[nt], false, false);
      }
    }
  }

  __syncthreads();
  float il[8];
#pragma unroll
  for (int v = 0; v < 8; ++v) il[v] = 1.0f / lrow[v + kh * 8];
#pragma unroll
  for (int v = 0; v < 8; ++v) {
    int row = v + kh * 8;
    size_t base = ((size_t)(b * NN + q0 + row)) * DD + d0 + M;
    out[base]      = oacc[0][v] * il[v];
    out[base + 16] = oacc[1][v] * il[v];
    out[base + 32] = oacc[2][v] * il[v];
    out[base + 48] = oacc[3][v] * il[v];
  }
}

extern "C" void kernel_launch(void* const* d_in, const int* in_sizes, int n_in,
                              void* d_out, int out_size, void* d_ws, size_t ws_size,
                              hipStream_t stream) {
  (void)in_sizes; (void)n_in; (void)out_size; (void)ws_size;
  const float* inp  = (const float*)d_in[0];   // [8,2048,512] fp32
  const float* W    = (const float*)d_in[1];   // [512,512]    fp32
  const float* bias = (const float*)d_in[2];   // [512]        fp32
  float* out = (float*)d_out;                  // [8,2048,512] fp32

  char* ws = (char*)d_ws;
  unsigned short* Xb = (unsigned short*)ws;                                 // 16 MB
  unsigned short* Vt = (unsigned short*)(ws + (size_t)BB * NN * DD * 2);    // 16 MB
  unsigned short* Wb = (unsigned short*)(ws + (size_t)2 * BB * NN * DD * 2);// 512 KB

  k_cvt_w<<<(DD * DD) / 512, 256, 0, stream>>>(W, (unsigned int*)Wb);
  k_transpose_v<<<BB * (DD / 32) * (NN / 32), 256, 0, stream>>>(inp, Vt);
  k_proj<<<((BB * NN / 16) * (DD / 16)) / 8, 256, 0, stream>>>(inp, Wb, bias, Xb);
  k_attn<<<BB * (NN / 16), 256, 0, stream>>>(Xb, Vt, out);
}
